// GptOssDecoderLayer_65738769433509
// MI455X (gfx1250) — matrix operand order repaired and checked by reading, not compiled, not run on hardware
//
#include <hip/hip_runtime.h>
#include <hip/hip_bf16.h>
#include <cstddef>
#include <cstdint>

// ---------------------------------------------------------------------------
// GPT-OSS decoder layer for MI455X (gfx1250, wave32, WMMA).
// All matmuls via v_wmma_f32_16x16x32_bf16 (fp32 accum). Activations that feed
// GEMM A-operands are kept in bf16 so A tiles can be staged into LDS with the
// CDNA5 async global->LDS path (ASYNCcnt) when the toolchain exposes it.
// ---------------------------------------------------------------------------

typedef __attribute__((ext_vector_type(16))) __bf16 bf16x16;
typedef __attribute__((ext_vector_type(8)))  __bf16 bf16x8;
typedef __attribute__((ext_vector_type(8)))  float  v8f;

#define S_LEN 1024
#define HDIM  2048
#define NHEAD 32
#define DHEAD 64
#define NEXP  8
#define IDIM  2048

static __device__ __forceinline__ bf16x16 cat8(bf16x8 a, bf16x8 b) {
  return __builtin_shufflevector(a, b, 0,1,2,3,4,5,6,7,8,9,10,11,12,13,14,15);
}

// ---- async global->LDS staging (gfx1250) with safe fallback ----------------
// Builtin signature (from hipcc diagnostic): param0 is
// 'int __vector(4) __device__ *'  -> vector_size(16) int, addrspace(1).
#if __has_builtin(__builtin_amdgcn_global_load_async_to_lds_b128)
#define HAS_ASYNC_LDS 1
typedef int v4i_gcc __attribute__((vector_size(16)));
typedef __attribute__((address_space(1))) v4i_gcc gl_v4i;
typedef __attribute__((address_space(3))) v4i_gcc lds_v4i;
#else
#define HAS_ASYNC_LDS 0
#endif

static __device__ __forceinline__ void stage_b128(const __bf16* gp, __bf16* lp) {
#if HAS_ASYNC_LDS
  __builtin_amdgcn_global_load_async_to_lds_b128((gl_v4i*)gp, (lds_v4i*)lp, 0, 0);
#else
  *(bf16x8*)lp = *(const bf16x8*)gp;
#endif
}

static __device__ __forceinline__ void stage_wait() {
#if HAS_ASYNC_LDS
#if __has_builtin(__builtin_amdgcn_s_wait_asynccnt)
  __builtin_amdgcn_s_wait_asynccnt(0);
#else
  asm volatile("s_wait_asynccnt 0" ::: "memory");
#endif
#endif
}

// ---------------------------------------------------------------------------
// RMSNorm (matches reference: out = w * x * sqrt(mean(x^2) + eps)) -> bf16
// ---------------------------------------------------------------------------
__global__ __launch_bounds__(256) void rmsnorm_kernel(
    const float* __restrict__ x, const float* __restrict__ w,
    __bf16* __restrict__ out) {
  const int s = blockIdx.x;
  const int tid = threadIdx.x;
  __shared__ float red[8];
  __shared__ float rtot;
  float ss = 0.f;
  for (int c = tid; c < HDIM; c += 256) {
    float v = x[(size_t)s * HDIM + c];
    ss += v * v;
  }
  for (int off = 16; off > 0; off >>= 1) ss += __shfl_xor(ss, off, 32);
  if ((tid & 31) == 0) red[tid >> 5] = ss;
  __syncthreads();
  if (tid == 0) {
    float t = 0.f;
    for (int i = 0; i < 8; ++i) t += red[i];
    rtot = sqrtf(t / (float)HDIM + 1e-5f);
  }
  __syncthreads();
  const float r = rtot;
  for (int c = tid; c < HDIM; c += 256)
    out[(size_t)s * HDIM + c] = (__bf16)(w[c] * x[(size_t)s * HDIM + c] * r);
}

// ---------------------------------------------------------------------------
// Tiled WMMA GEMM: C[M,N] = A_bf16[M,K] @ B + bias (+resid) (+rowscale-acc)
// BLAYOUT 0: B is W[N,K] row-major (q/k/v/o projections: h @ W^T)
// BLAYOUT 1: B is W[K,N] row-major (gate_up_w / down_w)
// Block: 256 threads (8 waves). Tile: 64(M) x 128(N) x 64(K).
// Wave w owns 32x32 output: M sub (w&1)*32, N sub (w>>1)*32 -> 4 accumulators,
// 8 WMMAs per K-stage. A tile staged via async global->LDS (bf16, no VALU).
// ---------------------------------------------------------------------------
template <int BLAYOUT>
__global__ __launch_bounds__(256) void gemm_kernel(
    const __bf16* __restrict__ A, const float* __restrict__ B,
    const float* __restrict__ bias, const float* __restrict__ resid,
    const float* __restrict__ rowscale, int rs_stride,
    float* __restrict__ C, int M, int N, int K) {
  constexpr int LDT = 80;  // bf16 row stride: 160B, 16B-aligned, conflict-padded
  __shared__ __bf16 sA[64 * LDT];
  __shared__ __bf16 sB[128 * LDT];

  const int tid  = threadIdx.x;
  const int lane = tid & 31;
  const int w    = tid >> 5;
  const int m0 = blockIdx.y * 64;
  const int n0 = blockIdx.x * 128;
  const int wm = (w & 1) * 32;
  const int wn = (w >> 1) * 32;
  const int mrA0 = wm + (lane & 15);       // A-fragment rows (two M sub-tiles)
  const int mrA1 = mrA0 + 16;
  const int kg   = (lane >> 4) * 8;        // A-fragment k-group
  const int nrB0 = wn + (lane & 15);       // B-fragment cols (two N sub-tiles)
  const int nrB1 = nrB0 + 16;
  const int kb   = (lane >> 4) * 16;       // unused (B fragments index with kg)

  v8f acc00 = {}, acc01 = {}, acc10 = {}, acc11 = {};

  for (int k0 = 0; k0 < K; k0 += 64) {
    __syncthreads();
    // --- stage A: 64x64 bf16, 16B per lane via async-to-LDS ---
#pragma unroll
    for (int it = 0; it < 2; ++it) {
      const int i = it * 256 + tid;
      const int r = i >> 3;
      const int c = (i & 7) * 8;
      stage_b128(A + (size_t)(m0 + r) * K + k0 + c, &sA[r * LDT + c]);
    }
    // --- stage B: 128x64, fp32 -> bf16 in-register ---
    if (BLAYOUT == 0) {
#pragma unroll
      for (int it = 0; it < 8; ++it) {
        const int i = it * 256 + tid;
        const int r = i >> 4;          // n (0..127)
        const int c = (i & 15) * 4;    // k (0..60)
        const float4 f = *(const float4*)(B + (size_t)(n0 + r) * K + k0 + c);
        __bf16* d = &sB[r * LDT + c];
        d[0] = (__bf16)f.x; d[1] = (__bf16)f.y; d[2] = (__bf16)f.z; d[3] = (__bf16)f.w;
      }
    } else {
#pragma unroll
      for (int it = 0; it < 8; ++it) {
        const int i = it * 256 + tid;
        const int r = i >> 5;          // k (0..63)
        const int c = (i & 31) * 4;    // n (0..124)
        const float4 f = *(const float4*)(B + (size_t)(k0 + r) * N + n0 + c);
        sB[(c + 0) * LDT + r] = (__bf16)f.x;
        sB[(c + 1) * LDT + r] = (__bf16)f.y;
        sB[(c + 2) * LDT + r] = (__bf16)f.z;
        sB[(c + 3) * LDT + r] = (__bf16)f.w;
      }
    }
    // prefetch next-stage B tile into cache (global_prefetch_b8)
    if (k0 + 64 < K) {
      if (BLAYOUT == 0)
        __builtin_prefetch(B + (size_t)(n0 + (tid >> 4)) * K + k0 + 64 + (tid & 15) * 4, 0, 2);
      else
        __builtin_prefetch(B + (size_t)(k0 + 64 + (tid >> 5)) * N + n0 + (tid & 31) * 4, 0, 2);
    }
    stage_wait();
    __syncthreads();

#pragma unroll
    for (int ks = 0; ks < 64; ks += 32) {
      bf16x16 a0 = cat8(*(const bf16x8*)&sA[mrA0 * LDT + ks + kg],
                        *(const bf16x8*)&sA[mrA0 * LDT + ks + 16 + kg]);
      bf16x16 a1 = cat8(*(const bf16x8*)&sA[mrA1 * LDT + ks + kg],
                        *(const bf16x8*)&sA[mrA1 * LDT + ks + 16 + kg]);
      bf16x16 b0 = cat8(*(const bf16x8*)&sB[nrB0 * LDT + ks + kg],
                        *(const bf16x8*)&sB[nrB0 * LDT + ks + 16 + kg]);
      bf16x16 b1 = cat8(*(const bf16x8*)&sB[nrB1 * LDT + ks + kg],
                        *(const bf16x8*)&sB[nrB1 * LDT + ks + 16 + kg]);
      acc00 = __builtin_amdgcn_wmma_f32_16x16x32_bf16(false, a0, false, b0, (short)0, acc00, false, false);
      acc01 = __builtin_amdgcn_wmma_f32_16x16x32_bf16(false, a0, false, b1, (short)0, acc01, false, false);
      acc10 = __builtin_amdgcn_wmma_f32_16x16x32_bf16(false, a1, false, b0, (short)0, acc10, false, false);
      acc11 = __builtin_amdgcn_wmma_f32_16x16x32_bf16(false, a1, false, b1, (short)0, acc11, false, false);
    }
  }

  // epilogue: C/D layout -> lane holds col (lane&15), rows r + 8*(lane>>4)
  const int gm0 = m0 + wm + 8 * (lane >> 4);
  const int gm1 = gm0 + 16;
  const int gn0 = n0 + wn + (lane & 15);
  const int gn1 = gn0 + 16;
#pragma unroll
  for (int r = 0; r < 8; ++r) {
    float v00 = acc00[r], v01 = acc01[r], v10 = acc10[r], v11 = acc11[r];
    if (bias) {
      v00 += bias[gn0]; v01 += bias[gn1];
      v10 += bias[gn0]; v11 += bias[gn1];
    }
    const int ra = gm0 + r, rb2 = gm1 + r;
    if (resid) {
      v00 += resid[(size_t)ra * N + gn0]; v01 += resid[(size_t)ra * N + gn1];
      v10 += resid[(size_t)rb2 * N + gn0]; v11 += resid[(size_t)rb2 * N + gn1];
    }
    const size_t ia0 = (size_t)ra * N + gn0, ia1 = (size_t)ra * N + gn1;
    const size_t ib0 = (size_t)rb2 * N + gn0, ib1 = (size_t)rb2 * N + gn1;
    if (rowscale) {
      const float sa = rowscale[(size_t)ra * rs_stride];
      const float sb = rowscale[(size_t)rb2 * rs_stride];
      C[ia0] += sa * v00; C[ia1] += sa * v01;
      C[ib0] += sb * v10; C[ib1] += sb * v11;
    } else {
      C[ia0] = v00; C[ia1] = v01;
      C[ib0] = v10; C[ib1] = v11;
    }
  }
}

// ---------------------------------------------------------------------------
// RoPE applied in-place to q or k laid out [S, NH*D] (fp32)
// ---------------------------------------------------------------------------
__global__ void rope_kernel(float* __restrict__ x, const float* __restrict__ cs,
                            const float* __restrict__ sn) {
  const int idx = blockIdx.x * blockDim.x + threadIdx.x;  // S*NH*32
  const int d  = idx & 31;
  const int hh = (idx >> 5) & (NHEAD - 1);
  const int s  = idx >> 10;
  const float c  = cs[s * 32 + d];
  const float si = sn[s * 32 + d];
  const size_t base = (size_t)s * HDIM + hh * DHEAD;
  const float a = x[base + d];
  const float b = x[base + 32 + d];
  x[base + d]      = a * c - b * si;
  x[base + 32 + d] = b * c + a * si;
}

// ---------------------------------------------------------------------------
// Flash attention with sink logits. Block = (64 q rows, 1 head), 8 waves.
// QK^T and PV through WMMA; online softmax fp32; sink folded into the final
// denominator. Output written bf16 (feeds the O-projection GEMM A-operand).
// ---------------------------------------------------------------------------
__global__ __launch_bounds__(256) void attn_kernel(
    const float* __restrict__ q, const float* __restrict__ k,
    const float* __restrict__ v, const float* __restrict__ sinks,
    __bf16* __restrict__ attn) {
  constexpr int LP = 72;  // bf16 stride (144B, 16B multiple)
  constexpr int LS = 66;  // f32 stride
  __shared__ __bf16 sQ[64 * LP];
  __shared__ __bf16 sK[64 * LP];
  __shared__ __bf16 sVt[64 * LP];  // transposed: [d][key]
  __shared__ float  sS[64 * LS];
  __shared__ __bf16 sP[64 * LP];
  __shared__ float rowM[64], rowS[64], rowF[64], rowN[64];

  const int tid  = threadIdx.x;
  const int lane = tid & 31;
  const int w    = tid >> 5;
  const int h    = blockIdx.y;
  const int q0   = blockIdx.x * 64;
  const size_t hoff = (size_t)h * DHEAD;

  const int tm  = (w & 3) * 16;
  const int tn  = (w >> 2) * 32;
  const int mr  = tm + (lane & 15);
  const int kg  = (lane >> 4) * 8;
  const int nr0 = tn + (lane & 15);
  const int nr1 = nr0 + 16;
  const int kb  = (lane >> 4) * 16;
  const int rb  = tm + 8 * (lane >> 4);

#pragma unroll
  for (int it = 0; it < 4; ++it) {
    int r = (tid >> 4) + it * 16;
    int c = (tid & 15) * 4;
    const float4 f = *(const float4*)(q + (size_t)(q0 + r) * HDIM + hoff + c);
    __bf16* d = &sQ[r * LP + c];
    d[0] = (__bf16)f.x; d[1] = (__bf16)f.y; d[2] = (__bf16)f.z; d[3] = (__bf16)f.w;
  }
  if (tid < 64) { rowM[tid] = -3.0e38f; rowS[tid] = 0.f; }

  v8f o0 = {}; v8f o1 = {};

  for (int kt = 0; kt <= (int)blockIdx.x; ++kt) {
    const int k0 = kt * 64;
    __syncthreads();
#pragma unroll
    for (int it = 0; it < 4; ++it) {
      int r = (tid >> 4) + it * 16;
      int c = (tid & 15) * 4;
      const float4 fk = *(const float4*)(k + (size_t)(k0 + r) * HDIM + hoff + c);
      __bf16* d = &sK[r * LP + c];
      d[0] = (__bf16)fk.x; d[1] = (__bf16)fk.y; d[2] = (__bf16)fk.z; d[3] = (__bf16)fk.w;
      const float4 fv = *(const float4*)(v + (size_t)(k0 + r) * HDIM + hoff + c);
      sVt[(c + 0) * LP + r] = (__bf16)fv.x;
      sVt[(c + 1) * LP + r] = (__bf16)fv.y;
      sVt[(c + 2) * LP + r] = (__bf16)fv.z;
      sVt[(c + 3) * LP + r] = (__bf16)fv.w;
    }
    __syncthreads();

    v8f s0 = {}; v8f s1 = {};
#pragma unroll
    for (int ks = 0; ks < DHEAD; ks += 32) {
      bf16x16 aF = cat8(*(const bf16x8*)&sQ[mr * LP + ks + kg],
                        *(const bf16x8*)&sQ[mr * LP + ks + 16 + kg]);
      bf16x16 b0 = cat8(*(const bf16x8*)&sK[nr0 * LP + ks + kg],
                        *(const bf16x8*)&sK[nr0 * LP + ks + 16 + kg]);
      bf16x16 b1 = cat8(*(const bf16x8*)&sK[nr1 * LP + ks + kg],
                        *(const bf16x8*)&sK[nr1 * LP + ks + 16 + kg]);
      s0 = __builtin_amdgcn_wmma_f32_16x16x32_bf16(false, aF, false, b0, (short)0, s0, false, false);
      s1 = __builtin_amdgcn_wmma_f32_16x16x32_bf16(false, aF, false, b1, (short)0, s1, false, false);
    }
#pragma unroll
    for (int r = 0; r < 8; ++r) {
      sS[(rb + r) * LS + tn + (lane & 15)]      = s0[r] * 0.125f;
      sS[(rb + r) * LS + tn + 16 + (lane & 15)] = s1[r] * 0.125f;
    }
    __syncthreads();

    if (tid < 64) {
      const int qg = q0 + tid;
      int kvalid = qg - k0 + 1;
      if (kvalid > 64) kvalid = 64;
      const float mo = rowM[tid];
      float mx = mo;
      for (int j = 0; j < kvalid; ++j) {
        const float sv = sS[tid * LS + j];
        if (sv > mx) mx = sv;
      }
      const float f = __expf(mo - mx);
      float sum = rowS[tid] * f;
      for (int j = 0; j < 64; ++j) {
        const float p = (j < kvalid) ? __expf(sS[tid * LS + j] - mx) : 0.f;
        sP[tid * LP + j] = (__bf16)p;
        sum += p;
      }
      rowM[tid] = mx; rowS[tid] = sum; rowF[tid] = f;
    }
    __syncthreads();

#pragma unroll
    for (int r = 0; r < 8; ++r) {
      const float f = rowF[rb + r];
      o0[r] *= f; o1[r] *= f;
    }
#pragma unroll
    for (int ks = 0; ks < 64; ks += 32) {
      bf16x16 aF = cat8(*(const bf16x8*)&sP[mr * LP + ks + kg],
                        *(const bf16x8*)&sP[mr * LP + ks + 16 + kg]);
      bf16x16 b0 = cat8(*(const bf16x8*)&sVt[nr0 * LP + ks + kg],
                        *(const bf16x8*)&sVt[nr0 * LP + ks + 16 + kg]);
      bf16x16 b1 = cat8(*(const bf16x8*)&sVt[nr1 * LP + ks + kg],
                        *(const bf16x8*)&sVt[nr1 * LP + ks + 16 + kg]);
      o0 = __builtin_amdgcn_wmma_f32_16x16x32_bf16(false, aF, false, b0, (short)0, o0, false, false);
      o1 = __builtin_amdgcn_wmma_f32_16x16x32_bf16(false, aF, false, b1, (short)0, o1, false, false);
    }
  }

  __syncthreads();
  if (tid < 64) {
    const float snk = sinks[h];
    const float m  = rowM[tid];
    const float mf = fmaxf(m, snk);
    const float denom = rowS[tid] * __expf(m - mf) + __expf(snk - mf);
    rowN[tid] = __expf(m - mf) / denom;
  }
  __syncthreads();
#pragma unroll
  for (int r = 0; r < 8; ++r) {
    const float nf = rowN[rb + r];
    attn[(size_t)(q0 + rb + r) * HDIM + hoff + nr0] = (__bf16)(o0[r] * nf);
    attn[(size_t)(q0 + rb + r) * HDIM + hoff + nr1] = (__bf16)(o1[r] * nf);
  }
}

// ---------------------------------------------------------------------------
// Router: one block per token, wave e computes logit for expert e; thread 0
// does top-2 + softmax -> dense score row (zeros elsewhere).
// ---------------------------------------------------------------------------
__global__ __launch_bounds__(256) void router_kernel(
    const __bf16* __restrict__ h, const float* __restrict__ rw,
    const float* __restrict__ rb, float* __restrict__ rsc) {
  const int t = blockIdx.x;
  const int tid = threadIdx.x;
  const int lane = tid & 31;
  const int e = tid >> 5;
  __shared__ float lg[NEXP];
  float s = 0.f;
  for (int c = lane; c < HDIM; c += 32)
    s += (float)h[(size_t)t * HDIM + c] * rw[(size_t)e * HDIM + c];
  for (int off = 16; off > 0; off >>= 1) s += __shfl_xor(s, off, 32);
  if (lane == 0) lg[e] = s + rb[e];
  __syncthreads();
  if (tid == 0) {
    int i0 = 0; float v0 = lg[0];
    for (int j = 1; j < NEXP; ++j) if (lg[j] > v0) { v0 = lg[j]; i0 = j; }
    int i1 = (i0 == 0) ? 1 : 0; float v1 = lg[i1];
    for (int j = 0; j < NEXP; ++j)
      if (j != i0 && lg[j] > v1) { v1 = lg[j]; i1 = j; }
    const float e1 = __expf(v1 - v0);
    const float inv = 1.0f / (1.0f + e1);
    for (int j = 0; j < NEXP; ++j) rsc[(size_t)t * NEXP + j] = 0.f;
    rsc[(size_t)t * NEXP + i0] = inv;
    rsc[(size_t)t * NEXP + i1] = e1 * inv;
  }
}

// ---------------------------------------------------------------------------
// Clamped SwiGLU: gate/up interleaved in gu[t][2i], gu[t][2i+1] -> bf16
// ---------------------------------------------------------------------------
__global__ void act_kernel(const float* __restrict__ gu, __bf16* __restrict__ act) {
  const int idx = blockIdx.x * blockDim.x + threadIdx.x;  // S*I
  const int t = idx >> 11;
  const int i = idx & (IDIM - 1);
  float g = gu[(size_t)t * (2 * IDIM) + 2 * i];
  float u = gu[(size_t)t * (2 * IDIM) + 2 * i + 1];
  g = fminf(g, 7.0f);
  u = fminf(fmaxf(u, -7.0f), 7.0f);
  const float sg = 1.0f / (1.0f + __expf(-1.702f * g));
  act[idx] = (__bf16)((u + 1.0f) * (g * sg));
}

__global__ void zero_kernel(float* __restrict__ p) {
  p[(size_t)blockIdx.x * blockDim.x + threadIdx.x] = 0.f;
}

__global__ void add_kernel(const float* __restrict__ a,
                           const float* __restrict__ b,
                           float* __restrict__ o) {
  const size_t i = (size_t)blockIdx.x * blockDim.x + threadIdx.x;
  o[i] = a[i] + b[i];
}

// ---------------------------------------------------------------------------
// Launch
// ---------------------------------------------------------------------------
extern "C" void kernel_launch(void* const* d_in, const int* in_sizes, int n_in,
                              void* d_out, int out_size, void* d_ws,
                              size_t ws_size, hipStream_t stream) {
  (void)in_sizes; (void)n_in; (void)out_size; (void)ws_size;
  const float* x     = (const float*)d_in[0];
  // d_in[1] = attention_mask (pure causal -1e9; applied analytically in attn_kernel)
  const float* cosv  = (const float*)d_in[2];
  const float* sinv  = (const float*)d_in[3];
  const float* wq    = (const float*)d_in[4];
  const float* bq    = (const float*)d_in[5];
  const float* wk    = (const float*)d_in[6];
  const float* bk    = (const float*)d_in[7];
  const float* wv    = (const float*)d_in[8];
  const float* bv    = (const float*)d_in[9];
  const float* wo    = (const float*)d_in[10];
  const float* bo    = (const float*)d_in[11];
  const float* sinks = (const float*)d_in[12];
  const float* ln1   = (const float*)d_in[13];
  const float* ln2   = (const float*)d_in[14];
  const float* rw    = (const float*)d_in[15];
  const float* rbias = (const float*)d_in[16];
  const float* guw   = (const float*)d_in[17];
  const float* gub   = (const float*)d_in[18];
  const float* dww   = (const float*)d_in[19];
  const float* dwb   = (const float*)d_in[20];
  float* out = (float*)d_out;

  // workspace carve-up (256B aligned slices)
  char* p = (char*)d_ws;
  const size_t T = (size_t)S_LEN * HDIM;  // 2M elements
  auto alloc = [&](size_t bytes) {
    char* r = p;
    p += (bytes + 255) & ~(size_t)255;
    return r;
  };
  __bf16* h1  = (__bf16*)alloc(T * 2);
  float*  qb  = (float*)alloc(T * 4);
  float*  kbf = (float*)alloc(T * 4);
  float*  vb  = (float*)alloc(T * 4);
  __bf16* at  = (__bf16*)alloc(T * 2);
  float*  x1  = (float*)alloc(T * 4);
  __bf16* h2  = (__bf16*)alloc(T * 2);
  float*  gu  = (float*)alloc(2 * T * 4);
  __bf16* acb = (__bf16*)alloc(T * 2);
  float*  moe = (float*)alloc(T * 4);
  float*  rsc = (float*)alloc((size_t)S_LEN * NEXP * 4);

  const dim3 gProj(HDIM / 128, S_LEN / 64);

  rmsnorm_kernel<<<S_LEN, 256, 0, stream>>>(x, ln1, h1);
  gemm_kernel<0><<<gProj, 256, 0, stream>>>(h1, wq, bq, nullptr, nullptr, 0, qb, S_LEN, HDIM, HDIM);
  gemm_kernel<0><<<gProj, 256, 0, stream>>>(h1, wk, bk, nullptr, nullptr, 0, kbf, S_LEN, HDIM, HDIM);
  gemm_kernel<0><<<gProj, 256, 0, stream>>>(h1, wv, bv, nullptr, nullptr, 0, vb, S_LEN, HDIM, HDIM);
  rope_kernel<<<(S_LEN * NHEAD * 32) / 256, 256, 0, stream>>>(qb, cosv, sinv);
  rope_kernel<<<(S_LEN * NHEAD * 32) / 256, 256, 0, stream>>>(kbf, cosv, sinv);
  attn_kernel<<<dim3(S_LEN / 64, NHEAD), 256, 0, stream>>>(qb, kbf, vb, sinks, at);
  gemm_kernel<0><<<gProj, 256, 0, stream>>>(at, wo, bo, x, nullptr, 0, x1, S_LEN, HDIM, HDIM);
  rmsnorm_kernel<<<S_LEN, 256, 0, stream>>>(x1, ln2, h2);
  router_kernel<<<S_LEN, 256, 0, stream>>>(h2, rw, rbias, rsc);
  zero_kernel<<<(unsigned)(T / 256), 256, 0, stream>>>(moe);

  for (int e = 0; e < NEXP; ++e) {
    gemm_kernel<1><<<dim3(2 * IDIM / 128, S_LEN / 64), 256, 0, stream>>>(
        h2, guw + (size_t)e * HDIM * 2 * IDIM, gub + (size_t)e * 2 * IDIM,
        nullptr, nullptr, 0, gu, S_LEN, 2 * IDIM, HDIM);
    act_kernel<<<(S_LEN * IDIM) / 256, 256, 0, stream>>>(gu, acb);
    gemm_kernel<1><<<dim3(HDIM / 128, S_LEN / 64), 256, 0, stream>>>(
        acb, dww + (size_t)e * IDIM * HDIM, dwb + (size_t)e * HDIM,
        nullptr, rsc + e, NEXP, moe, S_LEN, HDIM, IDIM);
  }

  add_kernel<<<(unsigned)(T / 256), 256, 0, stream>>>(x1, moe, out);
}
